// TemporalPooling_8323646620554
// MI455X (gfx1250) — matrix operand order repaired and checked by reading, not compiled, not run on hardware
//
#include <hip/hip_runtime.h>

// ---------------------------------------------------------------------------
// TemporalPooling for MI455X (gfx1250, wave32, WMMA)
//   embeds = input @ W + b ; segment-mean over (batch_i*48 + win_i) ; permute
// Strategy: bf16 WMMA GEMM (memory-bound @ 23.3 TB/s streaming the 512MB
// activations) + fp32 atomic scatter into L2-resident segment sums (101MB <
// 192MB L2) + LDS-transposed finalize with exact fp32 bias/divide.
// Read-once streams (input, indices, output, finalize-read of sums) use
// non-temporal hints so the segment-sum array keeps its L2 residency.
// ---------------------------------------------------------------------------

typedef __attribute__((ext_vector_type(16))) __bf16 v16bf;
typedef __attribute__((ext_vector_type(8)))  float  v8f;
typedef __attribute__((ext_vector_type(4)))  float  f32x4;

#define DIN   64
#define DOUT  128
#define WINSZ 48

union BF16Frag {
    v16bf v;
    unsigned short u[16];
    uint4 q[2];
};

__device__ __forceinline__ unsigned short f32_to_bf16_rne(float x) {
    unsigned int u = __float_as_uint(x);
    u = (u + 0x7FFFu + ((u >> 16) & 1u)) >> 16;   // round-to-nearest-even
    return (unsigned short)u;
}

// ---------------------------------------------------------------------------
// Kernel 1: zero sums + counts region of workspace (float4 stores)
// ---------------------------------------------------------------------------
__global__ void tp_zero_kernel(f32x4* __restrict__ p, int n4) {
    int i = blockIdx.x * blockDim.x + threadIdx.x;
    int stride = gridDim.x * blockDim.x;
    f32x4 z = {0.f, 0.f, 0.f, 0.f};
    for (; i < n4; i += stride) p[i] = z;
}

// ---------------------------------------------------------------------------
// Kernel 2: pack W (DIN x DOUT fp32, row-major) into 16 bf16 B-fragments in
// the per-lane layout this file uses for V_WMMA_F32_16X16X32_BF16 (contiguous k, matched by A):
//   frag f = ktile*8 + ntile  (ktile in {0,1} covers K 0..31 / 32..63)
//   lane L (<16): N = ntile*16 + L,     elems j=0..15 -> K = ktile*32 + j
//   lane L (>=16): same N base with L-16, K = ktile*32 + 16 + j
// Stored so each lane reads its 16 bf16 (32 bytes) contiguously.
// ---------------------------------------------------------------------------
__global__ void tp_pack_w_kernel(const float* __restrict__ W,
                                 unsigned short* __restrict__ bfrag) {
    int idx = blockIdx.x * blockDim.x + threadIdx.x;   // 0 .. 16*512-1
    if (idx >= 16 * 512) return;
    int f = idx >> 9;          // fragment
    int r = idx & 511;
    int L = r >> 4;            // lane 0..31
    int j = r & 15;            // element within lane
    int kt = f >> 3, nt = f & 7;
    int k = kt * 32 + ((L >> 4) * 16) + j;
    int n = nt * 16 + (L & 15);
    bfrag[idx] = f32_to_bf16_rne(W[k * DOUT + n]);
}

// ---------------------------------------------------------------------------
// Kernel 3: WMMA GEMM + atomic segment scatter.
// Each wave processes 16-event tiles (grid-stride over tiles). B fragments are
// register-resident for the lifetime of the wave. All scatter addressing is
// 32-bit (sums < 2^31 elements) so atomics use SGPR-base + VGPR-offset mode.
// ---------------------------------------------------------------------------
__global__ void __launch_bounds__(256)
tp_gemm_scatter_kernel(const float* __restrict__ input,
                       const int*   __restrict__ batch_i,
                       const int*   __restrict__ win_i,
                       const unsigned short* __restrict__ bfrag,
                       float* __restrict__ sums,
                       float* __restrict__ counts,
                       int n_tiles) {
    const int lane = threadIdx.x & 31;
    const int half = lane >> 4;          // 0: lanes 0-15, 1: lanes 16-31

    // Load all 16 B fragments into registers (32B per lane per fragment).
    // Regular (cached) loads: this 16KB table is shared by every wave.
    BF16Frag B[16];
    const uint4* bq = reinterpret_cast<const uint4*>(bfrag);
#pragma unroll
    for (int f = 0; f < 16; ++f) {
        B[f].q[0] = bq[f * 64 + lane * 2 + 0];
        B[f].q[1] = bq[f * 64 + lane * 2 + 1];
    }

    int wave_id = (int)((blockIdx.x * blockDim.x + threadIdx.x) >> 5);
    int n_waves = (int)((gridDim.x * blockDim.x) >> 5);

    for (int t = wave_id; t < n_tiles; t += n_waves) {
        const int row = t * 16 + (lane & 15);     // lanes 16-31 mirror rows 0-15

        // ---- Load A tile (non-temporal: read-once 512MB stream). Lane needs
        //      K chunks [half*16, +16) and [32 + half*16, +16). ----
        f32x4 af[8];
        const f32x4* arow =
            reinterpret_cast<const f32x4*>(input + (size_t)row * DIN);
#pragma unroll
        for (int c = 0; c < 4; ++c) {
            af[c * 2 + 0] = __builtin_nontemporal_load(arow + (c >> 1) * 8 + half * 4 + (c & 1) * 2 + 0);
            af[c * 2 + 1] = __builtin_nontemporal_load(arow + (c >> 1) * 8 + half * 4 + (c & 1) * 2 + 1);
        }

        // ---- Convert to the two 16x32 bf16 A fragments (v_cvt_pk_bf16_f32) --
        const float* ap = reinterpret_cast<const float*>(af);
        v16bf A0, A1;
#pragma unroll
        for (int j = 0; j < 16; ++j) {
            A0[j] = (__bf16)ap[j];        // K 0..31  slice
            A1[j] = (__bf16)ap[16 + j];   // K 32..63 slice
        }

        // ---- 16 WMMAs: 2 K-steps x 8 N-tiles, fp32 accumulation ----
        v8f acc[8];
#pragma unroll
        for (int nt = 0; nt < 8; ++nt) {
            v8f z = {};
            acc[nt] = __builtin_amdgcn_wmma_f32_16x16x32_bf16(
                false, A0, false, B[nt].v, (short)0, z, false, false);
            acc[nt] = __builtin_amdgcn_wmma_f32_16x16x32_bf16(
                false, A1, false, B[8 + nt].v, (short)0, acc[nt], false, false);
        }

        // ---- Segment ids for the 16 rows of this tile (read-once streams) ---
        int bi = __builtin_nontemporal_load(batch_i + row);
        int wi = __builtin_nontemporal_load(win_i + row);
        int seg_mine   = bi * WINSZ + wi;
        int segoff_mine = seg_mine * DOUT;     // pre-scaled element offset

        // ---- Scatter: D layout -> row m = half*8 + v, col n = nt*16 + lane&15
        //      32-bit offsets; nt*16 folds into the instruction ioffset. ----
#pragma unroll
        for (int v = 0; v < 8; ++v) {
            int m = half * 8 + v;
            unsigned idx = (unsigned)__shfl(segoff_mine, m, 32) + (lane & 15);
#pragma unroll
            for (int nt = 0; nt < 8; ++nt)
                unsafeAtomicAdd(sums + idx + nt * 16, acc[nt][v]);
        }

        // ---- Counts: one +1 per event (lanes 0-15 own distinct rows) ----
        if (half == 0) unsafeAtomicAdd(counts + (unsigned)seg_mine, 1.0f);
    }
}

// ---------------------------------------------------------------------------
// Kernel 4: scalar fp32 tail for N % 16 events (robustness; 0 for reference)
// ---------------------------------------------------------------------------
__global__ void tp_tail_kernel(const float* __restrict__ input,
                               const float* __restrict__ W,
                               const int* __restrict__ batch_i,
                               const int* __restrict__ win_i,
                               float* __restrict__ sums,
                               float* __restrict__ counts,
                               int start, int n) {
    int e = start + blockIdx.x * blockDim.x + threadIdx.x;
    if (e >= n) return;
    int seg = batch_i[e] * WINSZ + win_i[e];
    unsafeAtomicAdd(&counts[seg], 1.0f);
    const float* x = input + (size_t)e * DIN;
    for (int d = 0; d < DOUT; ++d) {
        float s = 0.f;
        for (int k = 0; k < DIN; ++k) s += x[k] * W[k * DOUT + d];
        unsafeAtomicAdd(&sums[(size_t)seg * DOUT + d], s);
    }
}

// ---------------------------------------------------------------------------
// Kernel 5: finalize. One block per batch element: divide by count, add bias,
// transpose [W,D] -> [D,W] through LDS (pitch 49: conflict-free both phases),
// coalesced global read and write. Sums are dead after this (NT load); output
// is write-once (NT store) -- keep L2 clean.
// ---------------------------------------------------------------------------
__global__ void __launch_bounds__(256)
tp_finalize_kernel(const float* __restrict__ sums,
                   const float* __restrict__ counts,
                   const float* __restrict__ bias,
                   float* __restrict__ out) {
    __shared__ float tile[DOUT * 49];
    const int b = blockIdx.x;
    const float* s = sums + (size_t)b * WINSZ * DOUT;
    const float* c = counts + (size_t)b * WINSZ;

    for (int j = threadIdx.x; j < WINSZ * DOUT; j += blockDim.x) {
        int w = j >> 7;          // DOUT == 128
        int d = j & (DOUT - 1);
        float cnt = c[w];
        float sv = __builtin_nontemporal_load(s + j);
        tile[d * 49 + w] = (cnt > 0.f) ? (sv / cnt + bias[d]) : 0.f;
    }
    __syncthreads();

    float* o = out + (size_t)b * DOUT * WINSZ;
    for (int j = threadIdx.x; j < DOUT * WINSZ; j += blockDim.x) {
        int d = j / WINSZ;
        int w = j - d * WINSZ;
        __builtin_nontemporal_store(tile[d * 49 + w], o + j);
    }
}

// ---------------------------------------------------------------------------
// Launch
//   inputs: 0 input[N,64] f32 | 1 W[64,128] f32 | 2 b[128] f32
//           3 batch_num (device scalar, derived from out_size instead)
//           4 batch_i[N] i32 | 5 win_i[N] i32
//   ws: sums[segs*128] f32 | counts[segs] f32 | bfrag[16*512] u16  (~101.5 MB)
// ---------------------------------------------------------------------------
extern "C" void kernel_launch(void* const* d_in, const int* in_sizes, int n_in,
                              void* d_out, int out_size, void* d_ws, size_t ws_size,
                              hipStream_t stream) {
    const float* input   = (const float*)d_in[0];
    const float* W       = (const float*)d_in[1];
    const float* bias    = (const float*)d_in[2];
    const int*   batch_i = (const int*)d_in[4];
    const int*   win_i   = (const int*)d_in[5];

    const int n_events = in_sizes[0] / DIN;
    const int batch    = out_size / (DOUT * WINSZ);
    const int segs     = batch * WINSZ;

    float* sums   = (float*)d_ws;
    float* counts = sums + (size_t)segs * DOUT;
    unsigned short* bfrag = (unsigned short*)(counts + segs);

    // 1. zero sums + counts (contiguous segs*(DOUT+1) floats, /4 exact)
    const int n4 = (segs * (DOUT + 1)) / 4;
    tp_zero_kernel<<<4096, 256, 0, stream>>>((f32x4*)d_ws, n4);

    // 2. pack W into bf16 WMMA B-fragments (16 KB, stays hot in L2/WGP$)
    tp_pack_w_kernel<<<32, 256, 0, stream>>>(W, bfrag);

    // 3. WMMA GEMM + atomic scatter (grid-stride over 16-event tiles)
    const int tiles = n_events / 16;
    if (tiles > 0)
        tp_gemm_scatter_kernel<<<2048, 256, 0, stream>>>(
            input, batch_i, win_i, bfrag, sums, counts, tiles);

    // 4. tail events (exact fp32 path)
    const int rem = n_events - tiles * 16;
    if (rem > 0)
        tp_tail_kernel<<<1, 64, 0, stream>>>(input, W, batch_i, win_i,
                                             sums, counts, tiles * 16, n_events);

    // 5. divide + bias + [B,W,D] -> [B,D,W] transpose
    tp_finalize_kernel<<<batch, 256, 0, stream>>>(sums, counts, bias,
                                                  (float*)d_out);
}
